// SynBertmodel_84911503442047
// MI455X (gfx1250) — matrix operand order repaired and checked by reading, not compiled, not run on hardware
//
#include <hip/hip_runtime.h>
#include <hip/hip_bf16.h>
#include <math.h>

// ---------------------------------------------------------------------------
// Types / helpers
// ---------------------------------------------------------------------------
typedef unsigned short ushort_t;
typedef __attribute__((ext_vector_type(16))) __bf16 v16bf;
typedef __attribute__((ext_vector_type(8)))  float  v8f;
typedef __attribute__((address_space(3))) ushort_t lds_ushort;

union Frag { uint4 q[2]; v16bf v; };

__device__ __forceinline__ unsigned lds_off_of(ushort_t* p) {
  return (unsigned)(size_t)(lds_ushort*)p;     // addrspace(3) ptr -> LDS byte offset
}

__device__ __forceinline__ ushort_t f2bf(float f) {
  unsigned int u = __float_as_uint(f);
  u += 0x7FFFu + ((u >> 16) & 1u);          // round-to-nearest-even
  return (ushort_t)(u >> 16);
}
__device__ __forceinline__ float bf2f(ushort_t h) {
  return __uint_as_float(((unsigned int)h) << 16);
}
__device__ __forceinline__ unsigned pack2bf(float a, float b) {
  return (unsigned)f2bf(a) | ((unsigned)f2bf(b) << 16);
}

// spans: widths 1..8, width w has 257-w spans, concatenated. S = 2020.
__device__ __forceinline__ int2 span_decode(int s) {
  int off = 0;
  #pragma unroll
  for (int w = 1; w <= 8; ++w) {
    int cnt = 257 - w;
    if (s < off + cnt) { int st = s - off; return make_int2(st, st + w); }
    off += cnt;
  }
  return make_int2(0, 1);
}

// searchsorted(bins, v, 'right') - 1
__device__ __forceinline__ int bucketOf(int v) {
  const int bins[14] = {0,1,2,3,4,5,7,8,15,16,31,32,63,64};
  int idx = -1;
  #pragma unroll
  for (int i = 0; i < 14; ++i) idx += (v >= bins[i]) ? 1 : 0;
  return idx;
}

// ---------------------------------------------------------------------------
// Problem constants
// ---------------------------------------------------------------------------
static constexpr int BATCH = 8, LSEQ = 256, DMODEL = 768, HID = 1024;
static constexpr int SSPAN = 2020;                 // spans per batch item
static constexpr int MSPAN = BATCH * SSPAN;        // 16160
static constexpr int MSPAN_PAD = 16256;            // 127 * 128
static constexpr int NZ = 64;
static constexpr int MPAIR = BATCH * NZ * NZ;      // 32768
static constexpr int KP1 = 1568;                   // 1561 padded to 32-mult

// ---------------------------------------------------------------------------
// Workspace layout (all sizes multiples of 256B; big region overlaid)
// ---------------------------------------------------------------------------
static constexpr size_t SZ_XBF   = 2048ull * 768 * 2;
static constexpr size_t SZ_WEMB  = 1024;
static constexpr size_t SZ_BTS1  = 2048ull * 768 * 2;
static constexpr size_t SZ_BTS2  = 1024ull * 1024 * 2;
static constexpr size_t SZ_BTP1  = 1024ull * KP1 * 2;
static constexpr size_t SZ_BTP2  = 1024ull * 1024 * 2;
static constexpr size_t SZ_H14   = 14ull * 1024 * 4;
static constexpr size_t SZ_IDX   = 8ull * 64 * 4;
static constexpr size_t SZ_APAIR = 1024ull * KP1 * 2;
static constexpr size_t SZ_HT    = 512ull * 1024 * 4;

static constexpr size_t OFF_XBF    = 0;
static constexpr size_t OFF_WEMB   = OFF_XBF   + SZ_XBF;
static constexpr size_t OFF_BTS1   = OFF_WEMB  + SZ_WEMB;
static constexpr size_t OFF_BTS2   = OFF_BTS1  + SZ_BTS1;
static constexpr size_t OFF_BTP1T  = OFF_BTS2  + SZ_BTS2;
static constexpr size_t OFF_BTP1O  = OFF_BTP1T + SZ_BTP1;
static constexpr size_t OFF_BTP2   = OFF_BTP1O + SZ_BTP1;
static constexpr size_t OFF_HW     = OFF_BTP2  + SZ_BTP2;
static constexpr size_t OFF_HD     = OFF_HW    + SZ_H14;
static constexpr size_t OFF_TIDX   = OFF_HD    + SZ_H14;
static constexpr size_t OFF_OIDX   = OFF_TIDX  + SZ_IDX;
static constexpr size_t OFF_APAIR  = OFF_OIDX  + SZ_IDX;
static constexpr size_t OFF_HT     = OFF_APAIR + SZ_APAIR;
static constexpr size_t OFF_HO     = OFF_HT    + SZ_HT;
static constexpr size_t OFF_BIG    = OFF_HO    + SZ_HT;
// span phase (dead before pair phase writes begin)
static constexpr size_t OFF_YZ     = OFF_BIG;
static constexpr size_t OFF_H1SPAN = OFF_YZ     + 2048ull * 2048 * 4;
static constexpr size_t OFF_H2SPAN = OFF_H1SPAN + (size_t)MSPAN_PAD * 1024 * 2;
// pair phase (overlays span phase)
static constexpr size_t OFF_H1PAIR = OFF_BIG;
static constexpr size_t OFF_H2PAIR = OFF_H1PAIR + (size_t)MPAIR * 1024 * 2;

// ---------------------------------------------------------------------------
// Conversion kernels
// ---------------------------------------------------------------------------
__global__ void k_f32_to_bf16(const float* __restrict__ in, ushort_t* __restrict__ out, int n) {
  int i = blockIdx.x * blockDim.x + threadIdx.x;
  if (i < n) out[i] = f2bf(in[i]);
}

// vectorized x4 (n must be a multiple of 4)
__global__ void k_f32_to_bf16_v4(const float* __restrict__ in, ushort_t* __restrict__ out, int n4) {
  int i = blockIdx.x * blockDim.x + threadIdx.x;
  if (i >= n4) return;
  float4 f = ((const float4*)in)[i];
  ((uint2*)out)[i] = make_uint2(pack2bf(f.x, f.y), pack2bf(f.z, f.w));
}

// out[n][k] = bf16(W[(rowoff+k)*ldw + n]) for k<Kreal else 0.
// LDS-tiled 32x32 transpose: coalesced reads along n, coalesced writes along k.
// grid (Kp/32, N/32), block (32,8). Requires N%32==0, Kp%32==0, ldw==N.
__global__ __launch_bounds__(256)
void k_wT_bf16(const float* __restrict__ W, int ldw, int rowoff, int Kreal, int Kp,
               ushort_t* __restrict__ out) {
  __shared__ float tile[32][33];
  const int k0 = blockIdx.x * 32;
  const int n0 = blockIdx.y * 32;
  const int tx = threadIdx.x;   // 0..31
  const int ty = threadIdx.y;   // 0..7
  #pragma unroll
  for (int s = 0; s < 4; ++s) {
    int k = k0 + ty + s * 8;
    float v = 0.f;
    if (k < Kreal) v = W[(size_t)(rowoff + k) * ldw + (n0 + tx)];
    tile[ty + s * 8][tx] = v;
  }
  __syncthreads();
  #pragma unroll
  for (int s = 0; s < 4; ++s) {
    int n = n0 + ty + s * 8;
    out[(size_t)n * Kp + (k0 + tx)] = f2bf(tile[tx][ty + s * 8]);
  }
}

// out[14][1024] = emb[14][25] @ W[rowoff:rowoff+25, :1024]
__global__ void k_emb_matmul(const float* __restrict__ emb, const float* __restrict__ W,
                             int ldw, int rowoff, float* __restrict__ out) {
  int i = blockIdx.x * blockDim.x + threadIdx.x;
  if (i >= 14 * 1024) return;
  int r = i / 1024, h = i % 1024;
  float acc = 0.f;
  #pragma unroll
  for (int j = 0; j < 25; ++j) acc += emb[r * 25 + j] * W[(size_t)(rowoff + j) * ldw + h];
  out[i] = acc;
}

// ---------------------------------------------------------------------------
// WMMA bf16 GEMM: C[M,N] = A[M,K] * Bt[N,K]^T (+bias, +relu)
// 128x128 tile, K-step 32, 256 threads = 8 waves (2x4), 8 WMMAs per wave/step.
// Double-buffered LDS filled by CDNA5 async DMA (global_load_async_to_lds_b128,
// ASYNCcnt-tracked): 1 barrier + 1 s_wait_asynccnt per K-step, DMA for step s+1
// overlaps ds_load+WMMA of step s.
// flags: bit0 relu, bit1 out bf16, bit2 add bias
// ---------------------------------------------------------------------------
#define LDSS 40   // LDS row stride in bf16 elems (80B, keeps b128 alignment)

#define ASYNC_B128(ldsoff, gaddr)                                              \
  asm volatile("global_load_async_to_lds_b128 %0, %1, off"                     \
               :: "v"(ldsoff), "v"(gaddr) : "memory")

__global__ __launch_bounds__(256)
void k_gemm(const ushort_t* __restrict__ A, const ushort_t* __restrict__ Bt,
            const float* __restrict__ bias, void* __restrict__ Cout,
            int M, int N, int K, int flags) {
  __shared__ ushort_t As[2][128 * LDSS];
  __shared__ ushort_t Bs[2][128 * LDSS];
  const int tid  = threadIdx.x;
  const int lane = tid & 31;
  const int wid  = tid >> 5;
  const int wm   = wid >> 2;     // 0..1  -> 64 rows each
  const int wn   = wid & 3;      // 0..3  -> 32 cols each
  const int half = lane >> 4;    // 0/1
  const int l16  = lane & 15;
  const int m0 = blockIdx.y * 128;
  const int n0 = blockIdx.x * 128;

  v8f acc[4][2];
  #pragma unroll
  for (int i = 0; i < 4; ++i)
    #pragma unroll
    for (int j = 0; j < 2; ++j)
      #pragma unroll
      for (int v = 0; v < 8; ++v) acc[i][j][v] = 0.f;

  // staging slots: thread covers (r0,q0) and (r0+64,q0) for both A and B tiles
  const int r0 = tid >> 2, q0 = tid & 3;
  const int r1 = r0 + 64;

  // global source pointers (element units; advance by 32 per K-step)
  const ushort_t* pA0 = A  + (size_t)(m0 + r0) * K + q0 * 8;
  const ushort_t* pA1 = A  + (size_t)(m0 + r1) * K + q0 * 8;
  const ushort_t* pB0 = Bt + (size_t)(n0 + r0) * K + q0 * 8;
  const ushort_t* pB1 = Bt + (size_t)(n0 + r1) * K + q0 * 8;

  // per-buffer LDS byte offsets for this thread's 4 DMA destinations
  unsigned la0[2], la1[2], lb0[2], lb1[2];
  #pragma unroll
  for (int b = 0; b < 2; ++b) {
    la0[b] = lds_off_of(&As[b][r0 * LDSS + q0 * 8]);
    la1[b] = lds_off_of(&As[b][r1 * LDSS + q0 * 8]);
    lb0[b] = lds_off_of(&Bs[b][r0 * LDSS + q0 * 8]);
    lb1[b] = lds_off_of(&Bs[b][r1 * LDSS + q0 * 8]);
  }

  const int nsteps = K >> 5;

  // prologue: DMA first K-slice into buffer 0
  {
    unsigned long long gA0 = (unsigned long long)pA0;
    unsigned long long gA1 = (unsigned long long)pA1;
    unsigned long long gB0 = (unsigned long long)pB0;
    unsigned long long gB1 = (unsigned long long)pB1;
    ASYNC_B128(la0[0], gA0);
    ASYNC_B128(la1[0], gA1);
    ASYNC_B128(lb0[0], gB0);
    ASYNC_B128(lb1[0], gB1);
  }

  for (int step = 0; step < nsteps; ++step) {
    const int buf = step & 1;
    // all of this wave's outstanding async DMAs target the current buffer
    asm volatile("s_wait_asynccnt 0" ::: "memory");
    __syncthreads();   // cur buffer visible to all; all waves done reading buf^1

    if (step + 1 < nsteps) {     // overlap next slice's DMA with this step's math
      size_t koff = (size_t)(step + 1) << 5;
      unsigned long long gA0 = (unsigned long long)(pA0 + koff);
      unsigned long long gA1 = (unsigned long long)(pA1 + koff);
      unsigned long long gB0 = (unsigned long long)(pB0 + koff);
      unsigned long long gB1 = (unsigned long long)(pB1 + koff);
      ASYNC_B128(la0[buf ^ 1], gA0);
      ASYNC_B128(la1[buf ^ 1], gA1);
      ASYNC_B128(lb0[buf ^ 1], gB0);
      ASYNC_B128(lb1[buf ^ 1], gB1);
    }

    Frag bf[2];
    #pragma unroll
    for (int j = 0; j < 2; ++j) {
      int n = wn * 32 + j * 16 + l16;
      // B (32x16): lane<16 holds K 0..15 (pairs per VGPR), lane>=16 holds K 16..31
      bf[j].q[0] = *(const uint4*)&Bs[buf][n * LDSS + half * 16];
      bf[j].q[1] = *(const uint4*)&Bs[buf][n * LDSS + half * 16 + 8];
    }
    #pragma unroll
    for (int i = 0; i < 4; ++i) {
      int m = wm * 64 + i * 16 + l16;
      // A (16x32): lane<16 holds K 0..15; lane>=16 holds K 16..31
      Frag af;
      af.q[0] = *(const uint4*)&As[buf][m * LDSS + half * 16];
      af.q[1] = *(const uint4*)&As[buf][m * LDSS + half * 16 + 8];
      #pragma unroll
      for (int j = 0; j < 2; ++j)
        acc[i][j] = __builtin_amdgcn_wmma_f32_16x16x32_bf16(
            false, af.v, false, bf[j].v, (short)0, acc[i][j], false, false);
    }
  }

  const bool doRelu = (flags & 1) != 0;
  const bool outBf  = (flags & 2) != 0;
  const bool hasB   = (flags & 4) != 0;
  #pragma unroll
  for (int i = 0; i < 4; ++i) {
    int rbase = m0 + wm * 64 + i * 16 + (half ? 8 : 0);   // C: VGPR v -> M = v (+8 hi lanes)
    #pragma unroll
    for (int j = 0; j < 2; ++j) {
      int col = n0 + wn * 32 + j * 16 + l16;
      float bv = hasB ? bias[col] : 0.f;
      #pragma unroll
      for (int v = 0; v < 8; ++v) {
        float val = acc[i][j][v] + bv;
        if (doRelu) val = fmaxf(val, 0.f);
        size_t off = (size_t)(rbase + v) * N + col;
        if (outBf) ((ushort_t*)Cout)[off] = f2bf(val);
        else       ((float*)Cout)[off]    = val;
      }
    }
  }
}

// ---------------------------------------------------------------------------
// h1_span[r,h] = relu(Y[b,start] + Z[b,end-1] + Hw[bucket(w)] + bs1); pad rows = 0
// YZ: [2048][2048] f32, cols 0..1023 = x@Ws1_a, cols 1024.. = x@Ws1_b
// 4 h-elements per thread (float4 / packed-bf16x4).
// ---------------------------------------------------------------------------
__global__ void k_h1_span(const float* __restrict__ YZ, const float* __restrict__ Hw,
                          const float* __restrict__ bs1, ushort_t* __restrict__ out) {
  size_t i = (size_t)blockIdx.x * blockDim.x + threadIdx.x;     // MSPAN_PAD*256
  if (i >= (size_t)MSPAN_PAD * 256) return;
  int r = (int)(i >> 8), h = (int)(i & 255) * 4;
  uint2 pk = make_uint2(0u, 0u);
  if (r < MSPAN) {
    int b = r / SSPAN, s = r % SSPAN;
    int2 se = span_decode(s);
    int bkt = bucketOf(se.y - se.x);
    float4 y  = *(const float4*)&YZ[(size_t)(b * LSEQ + se.x)     * 2048 + h];
    float4 z  = *(const float4*)&YZ[(size_t)(b * LSEQ + se.y - 1) * 2048 + 1024 + h];
    float4 hw = *(const float4*)&Hw[bkt * 1024 + h];
    float4 bb = *(const float4*)&bs1[h];
    float v0 = fmaxf(y.x + z.x + hw.x + bb.x, 0.f);
    float v1 = fmaxf(y.y + z.y + hw.y + bb.y, 0.f);
    float v2 = fmaxf(y.z + z.z + hw.z + bb.z, 0.f);
    float v3 = fmaxf(y.w + z.w + hw.w + bb.w, 0.f);
    pk = make_uint2(pack2bf(v0, v1), pack2bf(v2, v3));
  }
  ((uint2*)out)[i] = pk;
}

// ---------------------------------------------------------------------------
// span head: logits = h2 @ Ws3 + bs3; softmax(3) -> d_out[0:16160*3]
// ---------------------------------------------------------------------------
__global__ __launch_bounds__(128)
void k_span_head(const ushort_t* __restrict__ H2, const float* __restrict__ W3,
                 const float* __restrict__ b3, float* __restrict__ out) {
  __shared__ float w[1024 * 3];
  for (int i = threadIdx.x; i < 3072; i += blockDim.x) w[i] = W3[i];
  __syncthreads();
  int r = blockIdx.x * blockDim.x + threadIdx.x;
  if (r >= MSPAN) return;
  const ushort_t* hrow = H2 + (size_t)r * 1024;
  float a0 = b3[0], a1 = b3[1], a2 = b3[2];
  for (int k = 0; k < 1024; k += 8) {
    uint4 q = *(const uint4*)(hrow + k);
    unsigned int d[4] = {q.x, q.y, q.z, q.w};
    #pragma unroll
    for (int t = 0; t < 4; ++t) {
      float lo = __uint_as_float(d[t] << 16);
      float hi = __uint_as_float(d[t] & 0xFFFF0000u);
      int k0 = (k + t * 2) * 3, k1 = k0 + 3;
      a0 += lo * w[k0 + 0] + hi * w[k1 + 0];
      a1 += lo * w[k0 + 1] + hi * w[k1 + 1];
      a2 += lo * w[k0 + 2] + hi * w[k1 + 2];
    }
  }
  float m = fmaxf(a0, fmaxf(a1, a2));
  float e0 = __expf(a0 - m), e1 = __expf(a1 - m), e2 = __expf(a2 - m);
  float inv = 1.f / (e0 + e1 + e2);
  out[(size_t)r * 3 + 0] = e0 * inv;
  out[(size_t)r * 3 + 1] = e1 * inv;
  out[(size_t)r * 3 + 2] = e2 * inv;
}

// ---------------------------------------------------------------------------
// top-k (k=64) per (batch, channel): iterative argmax, ties -> lowest index
// (matches lax.top_k ordering). blockIdx.x = b*2 + (0:aspect ch1, 1:opinion ch2)
// ---------------------------------------------------------------------------
__global__ __launch_bounds__(256)
void k_topk(const float* __restrict__ span_prob, int* __restrict__ tidx, int* __restrict__ oidx) {
  __shared__ float vals[SSPAN];
  __shared__ float rv[256];
  __shared__ int   ri[256];
  int b = blockIdx.x >> 1;
  int c = (blockIdx.x & 1) + 1;
  int* outp = ((blockIdx.x & 1) == 0 ? tidx : oidx) + b * NZ;
  for (int s = threadIdx.x; s < SSPAN; s += blockDim.x)
    vals[s] = span_prob[(size_t)(b * SSPAN + s) * 3 + c];
  __syncthreads();
  for (int it = 0; it < NZ; ++it) {
    float bv = -INFINITY; int bi = 0x7FFFFFFF;
    for (int s = threadIdx.x; s < SSPAN; s += blockDim.x) {
      float v = vals[s];
      if (v > bv || (v == bv && s < bi)) { bv = v; bi = s; }
    }
    rv[threadIdx.x] = bv; ri[threadIdx.x] = bi;
    __syncthreads();
    for (int stride = 128; stride > 0; stride >>= 1) {
      if (threadIdx.x < stride) {
        float ov = rv[threadIdx.x + stride]; int oi = ri[threadIdx.x + stride];
        if (ov > rv[threadIdx.x] || (ov == rv[threadIdx.x] && oi < ri[threadIdx.x])) {
          rv[threadIdx.x] = ov; ri[threadIdx.x] = oi;
        }
      }
      __syncthreads();
    }
    if (threadIdx.x == 0) { outp[it] = ri[0]; vals[ri[0]] = -INFINITY; }
    __syncthreads();
  }
}

// ---------------------------------------------------------------------------
// Gather A_pair [1024][1568]: rows 0..511 = t spans, 512..1023 = o spans
// row layout: [x[start](768) | x[end-1](768) | width_emb(25) | pad(7)]
// ---------------------------------------------------------------------------
__global__ void k_gather_pairA(const ushort_t* __restrict__ xbf, const ushort_t* __restrict__ wembbf,
                               const int* __restrict__ tidx, const int* __restrict__ oidx,
                               ushort_t* __restrict__ A) {
  size_t i = (size_t)blockIdx.x * blockDim.x + threadIdx.x;
  if (i >= (size_t)1024 * KP1) return;
  int r = (int)(i / KP1), k = (int)(i % KP1);
  int b = (r >> 6) & 7, j = r & 63;
  int s = (r < 512) ? tidx[b * NZ + j] : oidx[b * NZ + j];
  int2 se = span_decode(s);
  ushort_t v = 0;
  if (k < 768)       v = xbf[(size_t)(b * LSEQ + se.x) * 768 + k];
  else if (k < 1536) v = xbf[(size_t)(b * LSEQ + se.y - 1) * 768 + (k - 768)];
  else if (k < 1561) v = wembbf[bucketOf(se.y - se.x) * 25 + (k - 1536)];
  A[i] = v;
}

// ---------------------------------------------------------------------------
// h1_pair[b,t,o,h] = relu(Ht[b,t,h] + Ho[b,o,h] + Hd[bucket(dist)][h] + bp1[h])
// dist = min(|end_t - start_o|, |start_t - end_o|). 4 h-elements per thread.
// ---------------------------------------------------------------------------
__global__ void k_h1_pair(const float* __restrict__ Ht, const float* __restrict__ Ho,
                          const float* __restrict__ Hd, const float* __restrict__ bp1,
                          const int* __restrict__ tidx, const int* __restrict__ oidx,
                          ushort_t* __restrict__ out) {
  size_t i = (size_t)blockIdx.x * blockDim.x + threadIdx.x;     // MPAIR*256
  if (i >= (size_t)MPAIR * 256) return;
  int r = (int)(i >> 8), h = (int)(i & 255) * 4;
  int b = r >> 12, t = (r >> 6) & 63, o = r & 63;
  int2 tse = span_decode(tidx[b * NZ + t]);
  int2 ose = span_decode(oidx[b * NZ + o]);
  int dist = min(abs(tse.y - ose.x), abs(tse.x - ose.y));
  int bkt = bucketOf(dist);
  float4 ht = *(const float4*)&Ht[(size_t)(b * NZ + t) * 1024 + h];
  float4 ho = *(const float4*)&Ho[(size_t)(b * NZ + o) * 1024 + h];
  float4 hd = *(const float4*)&Hd[bkt * 1024 + h];
  float4 bb = *(const float4*)&bp1[h];
  float v0 = fmaxf(ht.x + ho.x + hd.x + bb.x, 0.f);
  float v1 = fmaxf(ht.y + ho.y + hd.y + bb.y, 0.f);
  float v2 = fmaxf(ht.z + ho.z + hd.z + bb.z, 0.f);
  float v3 = fmaxf(ht.w + ho.w + hd.w + bb.w, 0.f);
  ((uint2*)out)[i] = make_uint2(pack2bf(v0, v1), pack2bf(v2, v3));
}

// ---------------------------------------------------------------------------
// pair head: logits = h2 @ Wp3 + bp3; softmax(4) -> d_out[48480:]
// ---------------------------------------------------------------------------
__global__ __launch_bounds__(128)
void k_pair_head(const ushort_t* __restrict__ H2, const float* __restrict__ W3,
                 const float* __restrict__ b3, float* __restrict__ out) {
  __shared__ float w[1024 * 4];
  for (int i = threadIdx.x; i < 4096; i += blockDim.x) w[i] = W3[i];
  __syncthreads();
  int r = blockIdx.x * blockDim.x + threadIdx.x;
  if (r >= MPAIR) return;
  const ushort_t* hrow = H2 + (size_t)r * 1024;
  float a0 = b3[0], a1 = b3[1], a2 = b3[2], a3 = b3[3];
  for (int k = 0; k < 1024; k += 8) {
    uint4 q = *(const uint4*)(hrow + k);
    unsigned int d[4] = {q.x, q.y, q.z, q.w};
    #pragma unroll
    for (int t = 0; t < 4; ++t) {
      float lo = __uint_as_float(d[t] << 16);
      float hi = __uint_as_float(d[t] & 0xFFFF0000u);
      int k0 = (k + t * 2) * 4, k1 = k0 + 4;
      a0 += lo * w[k0 + 0] + hi * w[k1 + 0];
      a1 += lo * w[k0 + 1] + hi * w[k1 + 1];
      a2 += lo * w[k0 + 2] + hi * w[k1 + 2];
      a3 += lo * w[k0 + 3] + hi * w[k1 + 3];
    }
  }
  float m = fmaxf(fmaxf(a0, a1), fmaxf(a2, a3));
  float e0 = __expf(a0 - m), e1 = __expf(a1 - m), e2 = __expf(a2 - m), e3 = __expf(a3 - m);
  float inv = 1.f / (e0 + e1 + e2 + e3);
  out[(size_t)r * 4 + 0] = e0 * inv;
  out[(size_t)r * 4 + 1] = e1 * inv;
  out[(size_t)r * 4 + 2] = e2 * inv;
  out[(size_t)r * 4 + 3] = e3 * inv;
}

// ---------------------------------------------------------------------------
// Host launcher
// ---------------------------------------------------------------------------
static inline unsigned cdiv(size_t a, unsigned b) { return (unsigned)((a + b - 1) / b); }

extern "C" void kernel_launch(void* const* d_in, const int* in_sizes, int n_in,
                              void* d_out, int out_size, void* d_ws, size_t ws_size,
                              hipStream_t stream) {
  (void)in_sizes; (void)n_in; (void)out_size; (void)ws_size;
  const float* x   = (const float*)d_in[0];
  const float* wE  = (const float*)d_in[1];
  const float* dE  = (const float*)d_in[2];
  const float* Ws1 = (const float*)d_in[3];
  const float* bs1 = (const float*)d_in[4];
  const float* Ws2 = (const float*)d_in[5];
  const float* bs2 = (const float*)d_in[6];
  const float* Ws3 = (const float*)d_in[7];
  const float* bs3 = (const float*)d_in[8];
  const float* Wp1 = (const float*)d_in[9];
  const float* bp1 = (const float*)d_in[10];
  const float* Wp2 = (const float*)d_in[11];
  const float* bp2 = (const float*)d_in[12];
  const float* Wp3 = (const float*)d_in[13];
  const float* bp3 = (const float*)d_in[14];

  char* ws = (char*)d_ws;
  ushort_t* xbf    = (ushort_t*)(ws + OFF_XBF);
  ushort_t* wembbf = (ushort_t*)(ws + OFF_WEMB);
  ushort_t* BtS1   = (ushort_t*)(ws + OFF_BTS1);
  ushort_t* BtS2   = (ushort_t*)(ws + OFF_BTS2);
  ushort_t* BtP1t  = (ushort_t*)(ws + OFF_BTP1T);
  ushort_t* BtP1o  = (ushort_t*)(ws + OFF_BTP1O);
  ushort_t* BtP2   = (ushort_t*)(ws + OFF_BTP2);
  float*    Hw     = (float*)   (ws + OFF_HW);
  float*    Hd     = (float*)   (ws + OFF_HD);
  int*      tidx   = (int*)     (ws + OFF_TIDX);
  int*      oidx   = (int*)     (ws + OFF_OIDX);
  ushort_t* Apair  = (ushort_t*)(ws + OFF_APAIR);
  float*    Ht     = (float*)   (ws + OFF_HT);
  float*    Ho     = (float*)   (ws + OFF_HO);
  float*    YZ     = (float*)   (ws + OFF_YZ);
  ushort_t* h1span = (ushort_t*)(ws + OFF_H1SPAN);
  ushort_t* h2span = (ushort_t*)(ws + OFF_H2SPAN);
  ushort_t* h1pair = (ushort_t*)(ws + OFF_H1PAIR);
  ushort_t* h2pair = (ushort_t*)(ws + OFF_H2PAIR);

  float* outSpan = (float*)d_out;                 // [16160,3]
  float* outPair = (float*)d_out + MSPAN * 3;     // [32768,4]

  // --- weight / input conversion (bf16, B pre-transposed to [N][K]) ---
  k_f32_to_bf16_v4<<<cdiv(2048ull*768/4, 256), 256, 0, stream>>>(x, xbf, 2048 * 768 / 4);
  k_f32_to_bf16<<<cdiv(14*25, 256), 256, 0, stream>>>(wE, wembbf, 14 * 25);
  k_wT_bf16<<<dim3( 768/32, 1024/32), dim3(32,8), 0, stream>>>(Ws1, 1024, 0,    768,  768,  BtS1);
  k_wT_bf16<<<dim3( 768/32, 1024/32), dim3(32,8), 0, stream>>>(Ws1, 1024, 768,  768,  768,  BtS1 + 1024ull*768);
  k_wT_bf16<<<dim3(1024/32, 1024/32), dim3(32,8), 0, stream>>>(Ws2, 1024, 0,    1024, 1024, BtS2);
  k_wT_bf16<<<dim3( KP1/32, 1024/32), dim3(32,8), 0, stream>>>(Wp1, 1024, 0,    1561, KP1,  BtP1t);
  k_wT_bf16<<<dim3( KP1/32, 1024/32), dim3(32,8), 0, stream>>>(Wp1, 1024, 1561, 1561, KP1,  BtP1o);
  k_wT_bf16<<<dim3(1024/32, 1024/32), dim3(32,8), 0, stream>>>(Wp2, 1024, 0,    1024, 1024, BtP2);
  k_emb_matmul<<<cdiv(14*1024, 256), 256, 0, stream>>>(wE, Ws1, 1024, 1536, Hw);
  k_emb_matmul<<<cdiv(14*1024, 256), 256, 0, stream>>>(dE, Wp1, 1024, 3122, Hd);

  // --- span path ---
  // YZ[2048,2048] = xbf[2048,768] @ [Ws1_a | Ws1_b]
  k_gemm<<<dim3(16, 16), 256, 0, stream>>>(xbf, BtS1, nullptr, YZ, 2048, 2048, 768, 0);
  k_h1_span<<<cdiv((size_t)MSPAN_PAD*256, 256), 256, 0, stream>>>(YZ, Hw, bs1, h1span);
  // h2 = relu(h1 @ Ws2 + bs2)
  k_gemm<<<dim3(8, MSPAN_PAD/128), 256, 0, stream>>>(h1span, BtS2, bs2, h2span, MSPAN_PAD, 1024, 1024, 7);
  k_span_head<<<cdiv(MSPAN, 128), 128, 0, stream>>>(h2span, Ws3, bs3, outSpan);
  k_topk<<<16, 256, 0, stream>>>(outSpan, tidx, oidx);

  // --- pair path ---
  k_gather_pairA<<<cdiv((size_t)1024*KP1, 256), 256, 0, stream>>>(xbf, wembbf, tidx, oidx, Apair);
  k_gemm<<<dim3(8, 4), 256, 0, stream>>>(Apair,                    BtP1t, nullptr, Ht, 512, 1024, KP1, 0);
  k_gemm<<<dim3(8, 4), 256, 0, stream>>>(Apair + 512ull*KP1,       BtP1o, nullptr, Ho, 512, 1024, KP1, 0);
  k_h1_pair<<<cdiv((size_t)MPAIR*256, 256), 256, 0, stream>>>(Ht, Ho, Hd, bp1, tidx, oidx, h1pair);
  k_gemm<<<dim3(8, MPAIR/128), 256, 0, stream>>>(h1pair, BtP2, bp2, h2pair, MPAIR, 1024, 1024, 7);
  k_pair_head<<<cdiv(MPAIR, 128), 128, 0, stream>>>(h2pair, Wp3, bp3, outPair);
}
